// TEDotProductAttention_59296318488990
// MI455X (gfx1250) — matrix operand order repaired and checked by reading, not compiled, not run on hardware
//
#include <hip/hip_runtime.h>

// ---------------------------------------------------------------------------
// Flash attention fwd, S=3072 B=1 H=16 D=64, fp32 in/out, bool mask.
// Transposed-score scheme: S^T = K * Q^T so softmax reductions are in-lane and
// P falls out directly in WMMA A-fragment order (no LDS restage, no shuffles).
// One-time preprocessing in d_ws: K -> f16 [h][s][d], V -> f16 [h][d][s],
// mask -> f16 additive bias [q][k] consumed as the WMMA C operand.
// ---------------------------------------------------------------------------

typedef __attribute__((ext_vector_type(16))) _Float16 v16h;
typedef __attribute__((ext_vector_type(8)))  _Float16 v8h;
typedef __attribute__((ext_vector_type(4)))  _Float16 v4h;
typedef __attribute__((ext_vector_type(8)))  float    v8f;

#define S_LEN  3072
#define NHEAD  16
#define DHEAD  64
#define KTILE  32                 // keys per inner iteration
#define NQ     2                  // query 16-tiles per wave
#define QK_SCALE 0.18033688011112042f  // (1/sqrt(64)) * log2(e)  -> exp2 domain
#define MASK_BIAS (-30000.0f)
#define NEG_BIG  (-1.0e30f)

#if __has_builtin(__builtin_amdgcn_exp2f)
#define EXP2F(x) __builtin_amdgcn_exp2f(x)
#else
#define EXP2F(x) exp2f(x)
#endif

static __device__ __forceinline__ v8f wmma_f16(v16h a, v16h b, v8f c) {
  return __builtin_amdgcn_wmma_f32_16x16x32_f16(
      false, a, false, b, (short)0, c, false, false);
}
static __device__ __forceinline__ v16h cat8(v8h lo, v8h hi) {
  return __builtin_shufflevector(lo, hi, 0,1,2,3,4,5,6,7,8,9,10,11,12,13,14,15);
}
// Broadcast with a pre-computed ds_bpermute byte address (hoists addr math).
static __device__ __forceinline__ float bcastf(float v, int byteaddr) {
  return __int_as_float(__builtin_amdgcn_ds_bpermute(byteaddr, __float_as_int(v)));
}

// ---------------------------------------------------------------------------
// Preprocessing kernels
// ---------------------------------------------------------------------------

// K f32 [s][h][d] -> f16 [h][s][d]
__global__ void prep_k_kernel(const float* __restrict__ K, _Float16* __restrict__ Kh) {
  int tid = blockIdx.x * 256 + threadIdx.x;       // 3072*16*16 threads
  int d4 = tid & 15, h = (tid >> 4) & 15, s = tid >> 8;
  float4 v = *(const float4*)(K + ((size_t)s * NHEAD + h) * DHEAD + d4 * 4);
  v4h o; o[0]=(_Float16)v.x; o[1]=(_Float16)v.y; o[2]=(_Float16)v.z; o[3]=(_Float16)v.w;
  *(v4h*)(Kh + ((size_t)h * S_LEN + s) * DHEAD + d4 * 4) = o;
}

// V f32 [s][h][d] -> f16 transposed [h][d][s]  (LDS-tiled transpose)
__global__ void prep_v_kernel(const float* __restrict__ V, _Float16* __restrict__ Vt) {
  __shared__ _Float16 tile[32][33];
  int h = blockIdx.z, d0 = blockIdx.y * 32, s0 = blockIdx.x * 32;
  int tx = threadIdx.x, ty = threadIdx.y;         // 32 x 8
#pragma unroll
  for (int i = 0; i < 4; ++i) {
    int sy = ty + i * 8;
    tile[sy][tx] = (_Float16)V[((size_t)(s0 + sy) * NHEAD + h) * DHEAD + d0 + tx];
  }
  __syncthreads();
#pragma unroll
  for (int i = 0; i < 4; ++i) {
    int dy = ty + i * 8;
    Vt[((size_t)h * DHEAD + d0 + dy) * S_LEN + s0 + tx] = tile[tx][dy];
  }
}

// bool mask [q][k] -> f16 additive bias [q][k]  (0 keep / -30000 drop)
__global__ void prep_bias_kernel(const unsigned char* __restrict__ M,
                                 _Float16* __restrict__ Bt) {
  int tid = blockIdx.x * 256 + threadIdx.x;       // S*S/4 threads
  uchar4 m4 = *(const uchar4*)(M + (size_t)tid * 4);
  v4h o;
  o[0] = m4.x ? (_Float16)0.0f : (_Float16)MASK_BIAS;
  o[1] = m4.y ? (_Float16)0.0f : (_Float16)MASK_BIAS;
  o[2] = m4.z ? (_Float16)0.0f : (_Float16)MASK_BIAS;
  o[3] = m4.w ? (_Float16)0.0f : (_Float16)MASK_BIAS;
  *(v4h*)(Bt + (size_t)tid * 4) = o;
}

// ---------------------------------------------------------------------------
// Main flash-attention kernel (preprocessed path)
// One wave32 per (head, pair of 16-row query tiles). Single-wave workgroups.
// ---------------------------------------------------------------------------
__launch_bounds__(32)
__global__ void fa_fwd_pre_kernel(const float* __restrict__ Q,
                                  const _Float16* __restrict__ Kh,  // [h][s][d]
                                  const _Float16* __restrict__ Vt,  // [h][d][s]
                                  const _Float16* __restrict__ Bt,  // [q][k] bias
                                  float* __restrict__ Out)
{
  const int lane = threadIdx.x;
  const int g  = lane >> 4;      // half-wave group
  const int ln = lane & 15;
  const int h  = blockIdx.x & (NHEAD - 1);
  const int qp = blockIdx.x >> 4;          // 0..95 (pair of query tiles)

  // Hoisted ds_bpermute byte addresses for per-q stat broadcast (row r+8g).
  int bpaddr[8];
#pragma unroll
  for (int r = 0; r < 8; ++r) bpaddr[r] = (r + 8 * g) << 2;

  // ---- Q^T B-fragments (N=q=ln, K=d), pre-scaled into exp2 domain ---------
  v16h qB[NQ][2];
#pragma unroll
  for (int iq = 0; iq < NQ; ++iq) {
    const int q = (qp * NQ + iq) * 16 + ln;
    const float* qb = Q + ((size_t)q * NHEAD + h) * DHEAD;
#pragma unroll
    for (int kk = 0; kk < 2; ++kk) {
      const float* bp = qb + kk * 32 + g * 16;   // 16 consecutive d values
#pragma unroll
      for (int i = 0; i < 16; ++i)
        qB[iq][kk][i] = (_Float16)(bp[i] * QK_SCALE);
    }
  }

  // ---- state -------------------------------------------------------------
  float mrow[NQ], lrow[NQ];
  v8f acc[NQ][4];                // O in C layout: M=q row r+8g, N=d dc*16+ln
#pragma unroll
  for (int iq = 0; iq < NQ; ++iq) {
    mrow[iq] = NEG_BIG; lrow[iq] = 0.0f;
#pragma unroll
    for (int dc = 0; dc < 4; ++dc) acc[iq][dc] = (v8f){0,0,0,0,0,0,0,0};
  }

  // ---- key loop ----------------------------------------------------------
  for (int t = 0; t < S_LEN / KTILE; ++t) {
    const int k0 = t * KTILE;

    // K A-fragments: M=k row (cb*16+ln), K-dim=d (kk selects d 0..31/32..63)
    v16h kA[2][2];
#pragma unroll
    for (int cb = 0; cb < 2; ++cb) {
      const _Float16* kr = Kh + ((size_t)h * S_LEN + k0 + cb * 16 + ln) * DHEAD;
#pragma unroll
      for (int kk = 0; kk < 2; ++kk) {
        const v8h* p = (const v8h*)(kr + kk * 32 + g * 16);
        kA[cb][kk] = cat8(p[0], p[1]);   // d offsets +0..15 contiguous
      }
    }

    // V B-fragments from V^T: N=d (dc*16+ln), K-dim=k (g*8+idx, +16 for idx>=8)
    v16h vB[4];
#pragma unroll
    for (int dc = 0; dc < 4; ++dc) {
      const _Float16* vr =
          Vt + ((size_t)h * DHEAD + dc * 16 + ln) * S_LEN + k0 + g * 8;
      vB[dc] = cat8(((const v8h*)vr)[0], ((const v8h*)vr)[2]);
    }

#pragma unroll
    for (int iq = 0; iq < NQ; ++iq) {
      const int q = (qp * NQ + iq) * 16 + ln;
      const _Float16* brow = Bt + (size_t)q * S_LEN + k0;

      // Scores S^T (M=k, N=q), mask folded in as the WMMA C operand.
      v8f s[2];
#pragma unroll
      for (int cb = 0; cb < 2; ++cb) {
        v8h bb = *(const v8h*)(brow + cb * 16 + g * 8);   // bias, k contiguous
        v8f c0;
#pragma unroll
        for (int r = 0; r < 8; ++r) c0[r] = (float)bb[r];
        s[cb] = wmma_f16(kA[cb][0], qB[iq][0], c0);
        s[cb] = wmma_f16(kA[cb][1], qB[iq][1], s[cb]);
      }

      // Online softmax for column q=ln: 16 in-lane values, pairwise trees,
      // one cross-half merge.
      float xs[16];
#pragma unroll
      for (int cb = 0; cb < 2; ++cb)
#pragma unroll
        for (int r = 0; r < 8; ++r) xs[cb * 8 + r] = s[cb][r];
#pragma unroll
      for (int w = 8; w >= 1; w >>= 1)
#pragma unroll
        for (int i = 0; i < w; ++i) xs[i] = fmaxf(xs[i], xs[i + w]);
      float mx = xs[0];
      mx = fmaxf(mx, __shfl_xor(mx, 16, 32));

      const float mold = mrow[iq];
      const float mnew = fmaxf(mold, mx);
      mrow[iq] = mnew;

      float p[2][8];
      float ps[16];
#pragma unroll
      for (int cb = 0; cb < 2; ++cb)
#pragma unroll
        for (int r = 0; r < 8; ++r) {
          p[cb][r] = EXP2F(s[cb][r] - mnew);
          ps[cb * 8 + r] = p[cb][r];
        }
#pragma unroll
      for (int w = 8; w >= 1; w >>= 1)
#pragma unroll
        for (int i = 0; i < w; ++i) ps[i] += ps[i + w];
      float sum = ps[0];
      sum += __shfl_xor(sum, 16, 32);

      // Rescale only when the running max actually grew (wave-uniform test).
      if (__any(mnew > mold)) {
        const float corr = EXP2F(mold - mnew);
        lrow[iq] = lrow[iq] * corr + sum;
#pragma unroll
        for (int r = 0; r < 8; ++r) {
          const float cr = bcastf(corr, bpaddr[r]);
#pragma unroll
          for (int dc = 0; dc < 4; ++dc) acc[iq][dc][r] *= cr;
        }
      } else {
        lrow[iq] += sum;
      }

      // P A-fragment (M=q=ln, K=k): pure in-lane repack of S^T C layout.
      v16h pa;
#pragma unroll
      for (int i = 0; i < 8; ++i) pa[i]     = (_Float16)p[0][i];
#pragma unroll
      for (int i = 0; i < 8; ++i) pa[8 + i] = (_Float16)p[1][i];

#pragma unroll
      for (int dc = 0; dc < 4; ++dc)
        acc[iq][dc] = wmma_f16(pa, vB[dc], acc[iq][dc]);
    }
  }

  // ---- finalize ----------------------------------------------------------
#pragma unroll
  for (int iq = 0; iq < NQ; ++iq) {
#pragma unroll
    for (int r = 0; r < 8; ++r) {
      const float li  = bcastf(lrow[iq], bpaddr[r]);
      const float inv = 1.0f / li;
      const size_t row =
          ((size_t)((qp * NQ + iq) * 16 + r + 8 * g) * NHEAD + h) * DHEAD;
#pragma unroll
      for (int dc = 0; dc < 4; ++dc)
        Out[row + dc * 16 + ln] = acc[iq][dc][r] * inv;
    }
  }
}

// ---------------------------------------------------------------------------
// Fallback (self-contained, no workspace): round-1 kernel.
// ---------------------------------------------------------------------------
static __device__ __forceinline__ float rowmax16(float v) {
#pragma unroll
  for (int off = 1; off < 16; off <<= 1) v = fmaxf(v, __shfl_xor(v, off, 16));
  return v;
}
static __device__ __forceinline__ float rowsum16(float v) {
#pragma unroll
  for (int off = 1; off < 16; off <<= 1) v += __shfl_xor(v, off, 16);
  return v;
}

__launch_bounds__(32)
__global__ void fa_fwd_fallback_kernel(const float* __restrict__ Q,
                                       const float* __restrict__ K,
                                       const float* __restrict__ V,
                                       const unsigned char* __restrict__ Mask,
                                       float* __restrict__ Out)
{
  __shared__ _Float16 pbuf[16 * 32];
  const int lane = threadIdx.x & 31;
  const int g = lane >> 4, ln = lane & 15;
  const int h = blockIdx.x & (NHEAD - 1);
  const int qt = blockIdx.x >> 4;

  v16h qf[2];
  {
    const float* qb = Q + ((size_t)(qt * 16 + ln) * NHEAD + h) * DHEAD;
#pragma unroll
    for (int kk = 0; kk < 2; ++kk)
#pragma unroll
      for (int blk = 0; blk < 2; ++blk) {
        const float* bp = qb + kk * 32 + blk * 8 + g * 16;
#pragma unroll
        for (int i = 0; i < 8; ++i)
          qf[kk][blk * 8 + i] = (_Float16)(bp[i] * 0.125f);
      }
  }

  float mrow[8], lrw[8];
  v8f acc[4];
#pragma unroll
  for (int r = 0; r < 8; ++r) { mrow[r] = NEG_BIG; lrw[r] = 0.0f; }
#pragma unroll
  for (int dc = 0; dc < 4; ++dc) acc[dc] = (v8f){0,0,0,0,0,0,0,0};

  for (int t = 0; t < S_LEN / 32; ++t) {
    const int k0 = t * 32;
    v16h kf[2][2];
#pragma unroll
    for (int cb = 0; cb < 2; ++cb) {
      const float* kr = K + ((size_t)(k0 + cb * 16 + ln) * NHEAD + h) * DHEAD;
#pragma unroll
      for (int kk = 0; kk < 2; ++kk) {
        const float* bp = kr + kk * 32 + g * 16;
#pragma unroll
        for (int i = 0; i < 16; ++i) kf[cb][kk][i] = (_Float16)bp[i];
      }
    }
    v8f s0 = (v8f){0,0,0,0,0,0,0,0}, s1 = (v8f){0,0,0,0,0,0,0,0};
    s0 = wmma_f16(qf[0], kf[0][0], s0); s0 = wmma_f16(qf[1], kf[0][1], s0);
    s1 = wmma_f16(qf[0], kf[1][0], s1); s1 = wmma_f16(qf[1], kf[1][1], s1);

    float x0[8], x1[8];
#pragma unroll
    for (int r = 0; r < 8; ++r) {
      const unsigned char* mr = Mask + (size_t)(qt * 16 + r + 8 * g) * S_LEN + k0;
      x0[r] = mr[ln] ? s0[r] : NEG_BIG;
      x1[r] = mr[16 + ln] ? s1[r] : NEG_BIG;
    }
    float corr[8], p0[8], p1[8];
#pragma unroll
    for (int r = 0; r < 8; ++r) {
      float tmax = rowmax16(fmaxf(x0[r], x1[r]));
      float mnew = fmaxf(mrow[r], tmax);
      corr[r] = __expf(mrow[r] - mnew); mrow[r] = mnew;
      p0[r] = __expf(x0[r] - mnew); p1[r] = __expf(x1[r] - mnew);
      lrw[r] = lrw[r] * corr[r] + rowsum16(p0[r] + p1[r]);
    }
#pragma unroll
    for (int dc = 0; dc < 4; ++dc)
#pragma unroll
      for (int r = 0; r < 8; ++r) acc[dc][r] *= corr[r];

#pragma unroll
    for (int r = 0; r < 8; ++r) {
      pbuf[(r + 8 * g) * 32 + ln] = (_Float16)p0[r];
      pbuf[(r + 8 * g) * 32 + 16 + ln] = (_Float16)p1[r];
    }
    __syncthreads();
    v16h pa;
#pragma unroll
    for (int idx = 0; idx < 16; ++idx)
      pa[idx] = pbuf[ln * 32 + g * 16 + idx];
    __syncthreads();

#pragma unroll
    for (int dc = 0; dc < 4; ++dc) {
      const float* vb = V + ((size_t)(k0 + g * 16) * NHEAD + h) * DHEAD + dc * 16 + ln;
      v16h vf;
#pragma unroll
      for (int idx = 0; idx < 16; ++idx)
        vf[idx] = (_Float16)vb[(size_t)idx * NHEAD * DHEAD];
      acc[dc] = wmma_f16(pa, vf, acc[dc]);
    }
  }
#pragma unroll
  for (int r = 0; r < 8; ++r) {
    float inv = 1.0f / lrw[r];
    size_t row = (size_t)(qt * 16 + r + 8 * g) * NHEAD + h;
#pragma unroll
    for (int dc = 0; dc < 4; ++dc)
      Out[row * DHEAD + dc * 16 + ln] = acc[dc][r] * inv;
  }
}

// ---------------------------------------------------------------------------
extern "C" void kernel_launch(void* const* d_in, const int* in_sizes, int n_in,
                              void* d_out, int out_size, void* d_ws, size_t ws_size,
                              hipStream_t stream) {
  (void)in_sizes; (void)n_in; (void)out_size;
  const float* Q = (const float*)d_in[0];
  const float* K = (const float*)d_in[1];
  const float* V = (const float*)d_in[2];
  const unsigned char* Mask = (const unsigned char*)d_in[3];  // jax bool = 1B
  float* Out = (float*)d_out;

  const size_t KH_BYTES = (size_t)NHEAD * S_LEN * DHEAD * 2;       // 6291456
  const size_t VT_BYTES = KH_BYTES;                                // 6291456
  const size_t BT_BYTES = (size_t)S_LEN * S_LEN * 2;               // 18874368
  const size_t NEED = KH_BYTES + VT_BYTES + BT_BYTES;              // ~30 MB

  if (ws_size >= NEED) {
    _Float16* Kh = (_Float16*)d_ws;
    _Float16* Vt = (_Float16*)((char*)d_ws + KH_BYTES);
    _Float16* Bt = (_Float16*)((char*)d_ws + KH_BYTES + VT_BYTES);

    prep_k_kernel<<<dim3(S_LEN * NHEAD * NHEAD / 256), dim3(256), 0, stream>>>(K, Kh);
    prep_v_kernel<<<dim3(S_LEN / 32, DHEAD / 32, NHEAD), dim3(32, 8), 0, stream>>>(V, Vt);
    prep_bias_kernel<<<dim3((size_t)S_LEN * S_LEN / 4 / 256), dim3(256), 0, stream>>>(Mask, Bt);

    fa_fwd_pre_kernel<<<dim3(NHEAD * (S_LEN / 16 / NQ)), dim3(32), 0, stream>>>(
        Q, Kh, Vt, Bt, Out);
  } else {
    fa_fwd_fallback_kernel<<<dim3(NHEAD * (S_LEN / 16)), dim3(32), 0, stream>>>(
        Q, K, V, Mask, Out);
  }
}
